// BestNet_46196668236142
// MI455X (gfx1250) — matrix operand order repaired and checked by reading, not compiled, not run on hardware
//
#include <hip/hip_runtime.h>

// ---------------------------------------------------------------- types
typedef _Float16 h4  __attribute__((ext_vector_type(4)));
typedef _Float16 h8  __attribute__((ext_vector_type(8)));
typedef _Float16 h16 __attribute__((ext_vector_type(16)));
typedef float    f4  __attribute__((ext_vector_type(4)));
typedef float    f8  __attribute__((ext_vector_type(8)));

#define LD      256          // d_model == d_state
#define NBATCH  32
#define SEQ     4096
#define BT      (NBATCH*SEQ) // 131072 rows
#define NCHUNK  64
#define TCHUNK  64           // NCHUNK*TCHUNK == SEQ
#define WMAT    65536        // 256*256 elements per weight matrix

// ---------------------------------------------------------------- WMMA helpers
__device__ __forceinline__ f8 wmma16(h16 a, h16 b, f8 c) {
  // v_wmma_f32_16x16x32_f16 ; f32 accumulate
  return __builtin_amdgcn_wmma_f32_16x16x32_f16(false, a, false, b, (short)0, c,
                                                false, false);
}

// A-matrix 16x32 f16 fragment: lane (m = L&15, kh = L>>4) holds 16 contiguous
// K values (kh*16 .. kh*16+15) in v[0..15], matching ldB.
__device__ __forceinline__ h16 ldA(const _Float16* p) {
  h8 lo = *(const h8*)p;
  h8 hi = *(const h8*)(p + 8);
  h16 r;
#pragma unroll
  for (int i = 0; i < 8; ++i) { r[i] = lo[i]; r[8 + i] = hi[i]; }
  return r;
}
// B-matrix 32x16 f16 fragment: lane (n = L&15, kh = L>>4) holds 16 contiguous
// K values (kh*16 .. kh*16+15) of column n == row n of the stored weight.
__device__ __forceinline__ h16 ldB(const _Float16* p) {
  h8 lo = *(const h8*)p;
  h8 hi = *(const h8*)(p + 8);
  h16 r;
#pragma unroll
  for (int i = 0; i < 8; ++i) { r[i] = lo[i]; r[8 + i] = hi[i]; }
  return r;
}

// ---------------------------------------------------------------- weight cvt
__global__ __launch_bounds__(256)
void kw_cvt(const float* __restrict__ bre, const float* __restrict__ bim,
            const float* __restrict__ cre, const float* __restrict__ cim,
            const float* __restrict__ dm,  const float* __restrict__ mw,
            _Float16* __restrict__ o) {
  int i = blockIdx.x * blockDim.x + threadIdx.x;
  o[i]          = (_Float16)bre[i];
  o[WMAT + i]   = (_Float16)bim[i];
  o[2*WMAT + i] = (_Float16)cre[i];
  o[3*WMAT + i] = (_Float16)(-cim[i]);   // pre-negate: f16 WMMA has no A/B neg
  o[4*WMAT + i] = (_Float16)dm[i];
  o[5*WMAT + i] = (_Float16)mw[i];
}

// ---------------------------------------------------------------- K1: LN1+leaky
__global__ __launch_bounds__(256)
void k1_ln1(const float* __restrict__ x,
            const float* __restrict__ g, const float* __restrict__ b,
            float* __restrict__ skip, _Float16* __restrict__ zact) {
  int row  = (blockIdx.x * blockDim.x + threadIdx.x) >> 5;   // one wave per row
  int lane = threadIdx.x & 31;
  const float* xr = x + (size_t)row * LD;
  f4 a0 = *(const f4*)(xr + lane * 4);
  f4 a1 = *(const f4*)(xr + 128 + lane * 4);
  float s = 0.f, s2 = 0.f;
#pragma unroll
  for (int i = 0; i < 4; ++i) {
    s += a0[i] + a1[i];
    s2 += a0[i]*a0[i] + a1[i]*a1[i];
  }
#pragma unroll
  for (int o = 1; o < 32; o <<= 1) {
    s  += __shfl_xor(s,  o, 32);
    s2 += __shfl_xor(s2, o, 32);
  }
  float mu = s * (1.f/256.f);
  float rs = rsqrtf(s2 * (1.f/256.f) - mu*mu + 1e-5f);
  f4 z0, z1; h4 q0, q1;
#pragma unroll
  for (int i = 0; i < 4; ++i) {
    int c0 = lane*4 + i, c1 = 128 + lane*4 + i;
    float v0 = (a0[i]-mu)*rs*g[c0] + b[c0];
    float v1 = (a1[i]-mu)*rs*g[c1] + b[c1];
    z0[i] = v0; z1[i] = v1;
    q0[i] = (_Float16)(v0 >= 0.f ? v0 : 0.01f*v0);
    q1[i] = (_Float16)(v1 >= 0.f ? v1 : 0.01f*v1);
  }
  *(f4*)(skip + (size_t)row*LD + lane*4)        = z0;
  *(f4*)(skip + (size_t)row*LD + 128 + lane*4)  = z1;
  *(h4*)(zact + (size_t)row*LD + lane*4)        = q0;
  *(h4*)(zact + (size_t)row*LD + 128 + lane*4)  = q1;
}

// ---------------------------------------------------------------- K2: u = γ·(B z)
// One wave per 16-row x 128-col half-strip, dual (re,im) accumulators:
// 2*8*8 = 128 accumulator VGPRs; launch_bounds lifts VGPR cap to 256.
__global__ __launch_bounds__(128, 1)
void k2_gemm_u(const _Float16* __restrict__ zact,
               const _Float16* __restrict__ wre,
               const _Float16* __restrict__ wim,
               const float* __restrict__ gamma_log,
               float* __restrict__ ure, float* __restrict__ uim) {
  int gwave = (blockIdx.x * blockDim.x + threadIdx.x) >> 5;
  int wave = gwave >> 1;          // 16-row strip index
  int half = gwave & 1;           // which 128-column half
  int lane = threadIdx.x & 31;
  int nl = lane & 15, kh = lane >> 4;
  size_t row0 = (size_t)wave * 16;
  f8 accR[8], accI[8];
#pragma unroll
  for (int nt = 0; nt < 8; ++nt)
#pragma unroll
    for (int e = 0; e < 8; ++e) { accR[nt][e] = 0.f; accI[nt][e] = 0.f; }

  const _Float16* arow = zact + (row0 + nl) * LD + kh * 16;
  const _Float16* wrh  = wre + (size_t)(half*128 + nl) * LD + kh * 16;
  const _Float16* wih  = wim + (size_t)(half*128 + nl) * LD + kh * 16;
  for (int kt = 0; kt < 8; ++kt) {
    h16 a = ldA(arow + kt * 32);
    if (kt < 7) __builtin_prefetch(arow + (kt + 1) * 32, 0, 1);
#pragma unroll
    for (int nt = 0; nt < 8; ++nt) {
      size_t wo = (size_t)nt * 16 * LD + kt * 32;
      accR[nt] = wmma16(a, ldB(wrh + wo), accR[nt]);
      accI[nt] = wmma16(a, ldB(wih + wo), accI[nt]);
    }
  }
#pragma unroll
  for (int nt = 0; nt < 8; ++nt) {
    int n = half*128 + nt*16 + nl;
    float gam = expf(gamma_log[n]);
#pragma unroll
    for (int r = 0; r < 8; ++r) {
      size_t o = (row0 + r + 8*kh) * LD + n;
      ure[o] = accR[nt][r] * gam;
      uim[o] = accI[nt][r] * gam;
    }
  }
}

// ---------------------------------------------------------------- scan (3 phases)
// h_t = (m_t * lam) * h_{t-1} + u_t ; product over a chunk = lam^TC * Π m.
__global__ __launch_bounds__(256)
void scanA(const float* __restrict__ ure, const float* __restrict__ uim,
           const unsigned char* __restrict__ done,
           const float* __restrict__ nu_log, const float* __restrict__ th_log,
           float* __restrict__ Hre, float* __restrict__ Him,
           float* __restrict__ Mall) {
  int n = threadIdx.x;
  int p = blockIdx.x % NCHUNK;
  int bb = blockIdx.x / NCHUNK;
  float labs = expf(-expf(nu_log[n]));
  float th   = expf(th_log[n]);
  float lr = labs * cosf(th), li = labs * sinf(th);
  float hr = 0.f, hi = 0.f, mprod = 1.f;
  size_t base = ((size_t)bb*SEQ + p*TCHUNK) * LD + n;
  const unsigned char* dn = done + (size_t)bb*SEQ + p*TCHUNK;
  for (int j = 0; j < TCHUNK; ++j) {
    float m = dn[j] ? 0.f : 1.f;
    mprod *= m;
    float pr = m*hr, pi = m*hi;
    hr = lr*pr - li*pi + ure[base + (size_t)j*LD];
    hi = lr*pi + li*pr + uim[base + (size_t)j*LD];
  }
  size_t o = ((size_t)bb*NCHUNK + p) * LD + n;
  Hre[o] = hr; Him[o] = hi; Mall[o] = mprod;
}

__global__ __launch_bounds__(256)
void scanB(const float* __restrict__ Hre, const float* __restrict__ Him,
           const float* __restrict__ Mall,
           const float* __restrict__ h0re, const float* __restrict__ h0im,
           const float* __restrict__ nu_log, const float* __restrict__ th_log,
           float* __restrict__ Pre, float* __restrict__ Pim) {
  int n = threadIdx.x, bb = blockIdx.x;
  float th   = expf(th_log[n]);
  float lA   = expf(-(float)TCHUNK * expf(nu_log[n]));   // |lam|^TC
  float ang  = (float)TCHUNK * th;
  float Lr = lA * cosf(ang), Li = lA * sinf(ang);
  float sr = h0re[bb*LD + n], si = h0im[bb*LD + n];
  for (int p = 0; p < NCHUNK; ++p) {
    size_t o = ((size_t)bb*NCHUNK + p) * LD + n;
    Pre[o] = sr; Pim[o] = si;
    float m = Mall[o];
    float pr = m*sr, pi = m*si;
    sr = Lr*pr - Li*pi + Hre[o];
    si = Lr*pi + Li*pr + Him[o];
  }
}

__global__ __launch_bounds__(256)
void scanC(const float* __restrict__ ure, const float* __restrict__ uim,
           const unsigned char* __restrict__ done,
           const float* __restrict__ nu_log, const float* __restrict__ th_log,
           const float* __restrict__ Pre, const float* __restrict__ Pim,
           _Float16* __restrict__ hsre, _Float16* __restrict__ hsim) {
  int n = threadIdx.x;
  int p = blockIdx.x % NCHUNK;
  int bb = blockIdx.x / NCHUNK;
  float labs = expf(-expf(nu_log[n]));
  float th   = expf(th_log[n]);
  float lr = labs * cosf(th), li = labs * sinf(th);
  size_t so = ((size_t)bb*NCHUNK + p) * LD + n;
  float hr = Pre[so], hi = Pim[so];
  size_t base = ((size_t)bb*SEQ + p*TCHUNK) * LD + n;
  const unsigned char* dn = done + (size_t)bb*SEQ + p*TCHUNK;
  for (int j = 0; j < TCHUNK; ++j) {
    float m = dn[j] ? 0.f : 1.f;
    float pr = m*hr, pi = m*hi;
    hr = lr*pr - li*pi + ure[base + (size_t)j*LD];
    hi = lr*pi + li*pr + uim[base + (size_t)j*LD];
    hsre[base + (size_t)j*LD] = (_Float16)hr;
    hsim[base + (size_t)j*LD] = (_Float16)hi;
  }
}

// -------------------------------------------- K4: y = hsR·Creᵀ − hsI·Cimᵀ + z·Dᵀ
//                                              then fused LN2 + leaky -> f16
__global__ __launch_bounds__(128, 1)
void k4_gemm_ln2(const _Float16* __restrict__ hsre,
                 const _Float16* __restrict__ hsim,
                 const _Float16* __restrict__ zact,
                 const _Float16* __restrict__ wc, // [Cre | -Cim | D] f16
                 const float* __restrict__ g2, const float* __restrict__ b2,
                 _Float16* __restrict__ yact) {
  int wave = (blockIdx.x * blockDim.x + threadIdx.x) >> 5;
  int lane = threadIdx.x & 31;
  int nl = lane & 15, kh = lane >> 4;
  size_t row0 = (size_t)wave * 16;
  f8 acc[16];
#pragma unroll
  for (int nt = 0; nt < 16; ++nt)
#pragma unroll
    for (int e = 0; e < 8; ++e) acc[nt][e] = 0.f;

  const _Float16* As[3] = {hsre, hsim, zact};
#pragma unroll
  for (int s = 0; s < 3; ++s) {
    const _Float16* A = As[s] + (row0 + nl) * LD + kh * 16;
    const _Float16* W = wc + (size_t)s * WMAT + (size_t)nl * LD + kh * 16;
    for (int kt = 0; kt < 8; ++kt) {
      h16 a = ldA(A + kt * 32);
      if (kt < 7) __builtin_prefetch(A + (kt + 1) * 32, 0, 1);
#pragma unroll
      for (int nt = 0; nt < 16; ++nt)
        acc[nt] = wmma16(a, ldB(W + (size_t)nt * 16 * LD + kt * 32), acc[nt]);
    }
  }
  // in-register LayerNorm over each of this lane-half's 8 rows
  float mu[8], rs[8];
#pragma unroll
  for (int r = 0; r < 8; ++r) {
    float s1 = 0.f, s2 = 0.f;
#pragma unroll
    for (int nt = 0; nt < 16; ++nt) { float v = acc[nt][r]; s1 += v; s2 += v*v; }
#pragma unroll
    for (int o = 1; o < 16; o <<= 1) {      // reduce within 16-lane half-wave
      s1 += __shfl_xor(s1, o, 32);
      s2 += __shfl_xor(s2, o, 32);
    }
    float m = s1 * (1.f/256.f);
    mu[r] = m; rs[r] = rsqrtf(s2 * (1.f/256.f) - m*m + 1e-5f);
  }
#pragma unroll
  for (int nt = 0; nt < 16; ++nt) {
    int n = nt*16 + nl;
    float gg = g2[n], bb = b2[n];
#pragma unroll
    for (int r = 0; r < 8; ++r) {
      float v = (acc[nt][r] - mu[r]) * rs[r] * gg + bb;
      v = v >= 0.f ? v : 0.01f * v;
      yact[(row0 + r + 8*kh) * LD + n] = (_Float16)v;
    }
  }
}

// -------------------------------------------- K6: out = LN3(y·mlp_wᵀ + b) + skip
__global__ __launch_bounds__(128, 1)
void k6_gemm_out(const _Float16* __restrict__ yact,
                 const _Float16* __restrict__ wm, // mlp_w f16
                 const float* __restrict__ bias,
                 const float* __restrict__ g3, const float* __restrict__ b3,
                 const float* __restrict__ skip,
                 float* __restrict__ out) {
  int wave = (blockIdx.x * blockDim.x + threadIdx.x) >> 5;
  int lane = threadIdx.x & 31;
  int nl = lane & 15, kh = lane >> 4;
  size_t row0 = (size_t)wave * 16;
  f8 acc[16];
#pragma unroll
  for (int nt = 0; nt < 16; ++nt)
#pragma unroll
    for (int e = 0; e < 8; ++e) acc[nt][e] = 0.f;

  const _Float16* A = yact + (row0 + nl) * LD + kh * 16;
  const _Float16* W = wm + (size_t)nl * LD + kh * 16;
  for (int kt = 0; kt < 8; ++kt) {
    h16 a = ldA(A + kt * 32);
    if (kt < 7) __builtin_prefetch(A + (kt + 1) * 32, 0, 1);
#pragma unroll
    for (int nt = 0; nt < 16; ++nt)
      acc[nt] = wmma16(a, ldB(W + (size_t)nt * 16 * LD + kt * 32), acc[nt]);
  }
  // + bias
#pragma unroll
  for (int nt = 0; nt < 16; ++nt) {
    float bz = bias[nt*16 + nl];
#pragma unroll
    for (int r = 0; r < 8; ++r) acc[nt][r] += bz;
  }
  // LN3
  float mu[8], rs[8];
#pragma unroll
  for (int r = 0; r < 8; ++r) {
    float s1 = 0.f, s2 = 0.f;
#pragma unroll
    for (int nt = 0; nt < 16; ++nt) { float v = acc[nt][r]; s1 += v; s2 += v*v; }
#pragma unroll
    for (int o = 1; o < 16; o <<= 1) {
      s1 += __shfl_xor(s1, o, 32);
      s2 += __shfl_xor(s2, o, 32);
    }
    float m = s1 * (1.f/256.f);
    mu[r] = m; rs[r] = rsqrtf(s2 * (1.f/256.f) - m*m + 1e-5f);
  }
#pragma unroll
  for (int nt = 0; nt < 16; ++nt) {
    int n = nt*16 + nl;
    float gg = g3[n], bb = b3[n];
#pragma unroll
    for (int r = 0; r < 8; ++r) {
      size_t o = (row0 + r + 8*kh) * LD + n;
      out[o] = (acc[nt][r] - mu[r]) * rs[r] * gg + bb + skip[o];
    }
  }
}

// ---------------------------------------------------------------- launcher
extern "C" void kernel_launch(void* const* d_in, const int* in_sizes, int n_in,
                              void* d_out, int out_size, void* d_ws, size_t ws_size,
                              hipStream_t stream) {
  (void)in_sizes; (void)n_in; (void)out_size; (void)ws_size;
  const float*         x       = (const float*)d_in[0];
  const unsigned char* done    = (const unsigned char*)d_in[1];
  const float*         h0re    = (const float*)d_in[2];
  const float*         h0im    = (const float*)d_in[3];
  const float*         nu_log  = (const float*)d_in[4];
  const float*         th_log  = (const float*)d_in[5];
  const float*         gam_log = (const float*)d_in[6];
  const float*         B_re    = (const float*)d_in[7];
  const float*         B_im    = (const float*)d_in[8];
  const float*         C_re    = (const float*)d_in[9];
  const float*         C_im    = (const float*)d_in[10];
  const float*         D_mat   = (const float*)d_in[11];
  const float*         mlp_w   = (const float*)d_in[12];
  const float*         mlp_b   = (const float*)d_in[13];
  const float*         ln1_g   = (const float*)d_in[14];
  const float*         ln1_b   = (const float*)d_in[15];
  const float*         ln2_g   = (const float*)d_in[16];
  const float*         ln2_b   = (const float*)d_in[17];
  const float*         ln3_g   = (const float*)d_in[18];
  const float*         ln3_b   = (const float*)d_in[19];

  char* ws = (char*)d_ws;
  const size_t SZ32 = (size_t)BT * LD * sizeof(float);   // 128 MB
  const size_t SZ16 = SZ32 / 2;
  float*    skip = (float*)ws;      ws += SZ32;
  _Float16* zact = (_Float16*)ws;   ws += SZ16;
  float*    ure  = (float*)ws;      ws += SZ32;
  float*    uim  = (float*)ws;      ws += SZ32;
  _Float16* hsre = (_Float16*)ws;   ws += SZ16;
  _Float16* hsim = (_Float16*)ws;   ws += SZ16;
  _Float16* w16  = (_Float16*)ws;   ws += (size_t)6 * WMAT * sizeof(_Float16);
  const size_t SUM = (size_t)NBATCH * NCHUNK * LD * sizeof(float); // 2 MB
  float* Hre  = (float*)ws; ws += SUM;
  float* Him  = (float*)ws; ws += SUM;
  float* Mall = (float*)ws; ws += SUM;
  float* Pre  = (float*)ws; ws += SUM;
  float* Pim  = (float*)ws; ws += SUM;
  _Float16* yact = (_Float16*)ure;  // u buffers dead after scanC -> reuse

  kw_cvt   <<<WMAT/256, 256, 0, stream>>>(B_re, B_im, C_re, C_im, D_mat, mlp_w, w16);
  k1_ln1   <<<BT/8, 256, 0, stream>>>(x, ln1_g, ln1_b, skip, zact);
  k2_gemm_u<<<(BT/16)*2/4, 128, 0, stream>>>(zact, w16, w16 + WMAT, gam_log,
                                             ure, uim);
  scanA    <<<NBATCH*NCHUNK, LD, 0, stream>>>(ure, uim, done, nu_log, th_log,
                                              Hre, Him, Mall);
  scanB    <<<NBATCH, LD, 0, stream>>>(Hre, Him, Mall, h0re, h0im, nu_log, th_log,
                                       Pre, Pim);
  scanC    <<<NBATCH*NCHUNK, LD, 0, stream>>>(ure, uim, done, nu_log, th_log,
                                              Pre, Pim, hsre, hsim);
  k4_gemm_ln2<<<(BT/16)/4, 128, 0, stream>>>(hsre, hsim, zact, w16 + 2*WMAT,
                                             ln2_g, ln2_b, yact);
  k6_gemm_out<<<(BT/16)/4, 128, 0, stream>>>(yact, w16 + 5*WMAT, mlp_b,
                                             ln3_g, ln3_b, skip, (float*)d_out);
}
